// DecoderNTM_28827820491520
// MI455X (gfx1250) — compile-verified, matrix operand order checked by reading, not run on hardware
//
#include <hip/hip_runtime.h>

#define DEV __device__ __forceinline__

typedef __bf16 bf16_t;
typedef __attribute__((ext_vector_type(8)))  bf16_t v8bf;
typedef __attribute__((ext_vector_type(16))) bf16_t v16bf;
typedef __attribute__((ext_vector_type(8)))  float  v8f;

// ---------------- problem constants ----------------
constexpr int T_STEPS = 128;
constexpr int BSZ     = 256;
constexpr int IDIM    = 256;
constexpr int CDIM    = 512;
constexpr int NMEM    = 128;   // memory slots
constexpr int MMEM    = 64;    // memory width
constexpr int KTOT    = IDIM + MMEM + CDIM;  // 832  ([emb | read | h])
constexpr int G4      = 4 * CDIM;            // 2048 (gate rows)
constexpr int RPAD    = 80;                  // read head cols padded (70 valid)
constexpr int WPAD    = 208;                 // write head cols padded (198 valid)
constexpr int NPROJ   = RPAD + WPAD;         // 288
constexpr int HOUT    = CDIM + MMEM;         // 576 output width per step

// ---------------- small math helpers ----------------
DEV float sigmoidf_(float x) { return 1.0f / (1.0f + expf(-x)); }
DEV float softplusf_(float x) { return (x > 20.0f) ? x : log1pf(expf(x)); }

// ---------------- WMMA fragment load ----------------
// bf16 16x32 A (and mirrored B) layout: lane&15 = row; lanes 0-15 hold
// K = [kb..kb+7] and [kb+16..kb+23] with kb = kk*32; lanes 16-31 the +8 halves.
// `p` points at rowBase + kk*32 + ((lane>>4)*8); both 16-byte chunks aligned.
DEV v16bf load_frag16(const bf16_t* p) {
    v8bf lo = *(const v8bf*)(p);
    v8bf hi = *(const v8bf*)(p + 16);
    v16bf f;
#pragma unroll
    for (int i = 0; i < 8; ++i) { f[i] = lo[i]; f[i + 8] = hi[i]; }
    return f;
}

// ================= init kernels =================
__global__ void k_init_Wc(const float* __restrict__ W_ih, const float* __restrict__ W_hh,
                          bf16_t* __restrict__ Wc) {
    int idx = blockIdx.x * blockDim.x + threadIdx.x;
    if (idx >= G4 * KTOT) return;
    int n = idx / KTOT, k = idx % KTOT;
    float v = (k < IDIM + MMEM) ? W_ih[(size_t)n * (IDIM + MMEM) + k]
                                : W_hh[(size_t)n * CDIM + (k - (IDIM + MMEM))];
    Wc[idx] = (bf16_t)v;
}

__global__ void k_init_WH(const float* __restrict__ read_W, const float* __restrict__ write_W,
                          bf16_t* __restrict__ WH) {
    int idx = blockIdx.x * blockDim.x + threadIdx.x;
    if (idx >= NPROJ * CDIM) return;
    int n = idx / CDIM, k = idx % CDIM;
    float v = 0.0f;
    if (n < RPAD) { if (n < 70) v = read_W[(size_t)k * 70 + n]; }
    else { int nn = n - RPAD; if (nn < 198) v = write_W[(size_t)k * 198 + nn]; }
    WH[idx] = (bf16_t)v;
}

__global__ void k_init_bias(const float* __restrict__ b_ih, const float* __restrict__ b_hh,
                            const float* __restrict__ read_b, const float* __restrict__ write_b,
                            float* __restrict__ bc, float* __restrict__ bh) {
    int idx = blockIdx.x * blockDim.x + threadIdx.x;
    if (idx < G4) bc[idx] = b_ih[idx] + b_hh[idx];
    if (idx < NPROJ) {
        float v = 0.0f;
        if (idx < 70) v = read_b[idx];
        else if (idx >= RPAD && idx - RPAD < 198) v = write_b[idx - RPAD];
        bh[idx] = v;
    }
}

__global__ void k_init_mem(const float* __restrict__ mem_bias, float* __restrict__ mem) {
    int idx = blockIdx.x * blockDim.x + threadIdx.x;
    if (idx >= BSZ * NMEM * MMEM) return;
    mem[idx] = mem_bias[idx % (NMEM * MMEM)];
}

__global__ void k_init_ch(const float* __restrict__ hid, const float* __restrict__ c0,
                          float* __restrict__ c, bf16_t* __restrict__ xh) {
    int idx = blockIdx.x * blockDim.x + threadIdx.x;
    if (idx >= BSZ * CDIM) return;
    int b = idx / CDIM, j = idx % CDIM;
    c[idx] = c0[j];
    xh[(size_t)b * KTOT + (IDIM + MMEM) + j] = (bf16_t)hid[idx];
}

__global__ void k_init_reads(const float* __restrict__ r0, bf16_t* __restrict__ xh) {
    int idx = blockIdx.x * blockDim.x + threadIdx.x;
    if (idx >= BSZ * MMEM) return;
    int b = idx / MMEM, m = idx % MMEM;
    xh[(size_t)b * KTOT + IDIM + m] = (bf16_t)r0[m];
}

__global__ void k_init_wst(float* __restrict__ wst) {
    int idx = blockIdx.x * blockDim.x + threadIdx.x;
    if (idx < 2 * BSZ * NMEM) wst[idx] = 0.0f;
}

// ================= per-step kernels =================
__global__ void k_copy_inp(const float* __restrict__ inp_t, bf16_t* __restrict__ xh) {
    int idx = blockIdx.x * blockDim.x + threadIdx.x;
    if (idx >= BSZ * IDIM) return;
    int b = idx / IDIM, j = idx % IDIM;
    xh[(size_t)b * KTOT + j] = (bf16_t)inp_t[idx];
}

// gates = xh(256x832) @ Wc^T(832x2048) + bc ; 8 waves/block, wave = 16x16 tile
__global__ void k_gemm_gates(const bf16_t* __restrict__ xh, const bf16_t* __restrict__ Wc,
                             const float* __restrict__ bc, float* __restrict__ gates) {
    const int lane = threadIdx.x & 31;
    const int wave = threadIdx.x >> 5;
    const int m0 = (blockIdx.y * 2 + (wave >> 2)) * 16;   // gridDim.y = 8
    const int n0 = (blockIdx.x * 4 + (wave & 3)) * 16;    // gridDim.x = 32
    const int r = lane & 15;
    const int khalf = (lane >> 4) * 8;
    const bf16_t* arow = xh + (size_t)(m0 + r) * KTOT;
    const bf16_t* brow = Wc + (size_t)(n0 + r) * KTOT;
    v8f acc = {};
#pragma unroll 2
    for (int kk = 0; kk < KTOT / 32; ++kk) {
        v16bf a = load_frag16(arow + kk * 32 + khalf);
        v16bf b = load_frag16(brow + kk * 32 + khalf);
        acc = __builtin_amdgcn_wmma_f32_16x16x32_bf16(false, a, false, b, (short)0, acc,
                                                      false, false);
    }
    const int col = n0 + r;
    const int rowbase = m0 + khalf;  // D layout: VGPR j -> M = j (+8 for lanes>=16)
    const float bias = bc[col];
#pragma unroll
    for (int j = 0; j < 8; ++j)
        gates[(size_t)(rowbase + j) * G4 + col] = acc[j] + bias;
}

// oproj = h(256x512) @ WH^T(512x288) + bh ; h read from xh columns 320..831
__global__ void k_gemm_proj(const bf16_t* __restrict__ xh, const bf16_t* __restrict__ WH,
                            const float* __restrict__ bh, float* __restrict__ oproj) {
    const int lane = threadIdx.x & 31;
    const int wave = threadIdx.x >> 5;                    // 6 waves/block
    const int m0 = blockIdx.y * 16;                       // gridDim.y = 16
    const int n0 = (blockIdx.x * 6 + wave) * 16;          // gridDim.x = 3
    const int r = lane & 15;
    const int khalf = (lane >> 4) * 8;
    const bf16_t* arow = xh + (size_t)(m0 + r) * KTOT + (IDIM + MMEM);
    const bf16_t* brow = WH + (size_t)(n0 + r) * CDIM;
    v8f acc = {};
#pragma unroll 2
    for (int kk = 0; kk < CDIM / 32; ++kk) {
        v16bf a = load_frag16(arow + kk * 32 + khalf);
        v16bf b = load_frag16(brow + kk * 32 + khalf);
        acc = __builtin_amdgcn_wmma_f32_16x16x32_bf16(false, a, false, b, (short)0, acc,
                                                      false, false);
    }
    const int col = n0 + r;
    const int rowbase = m0 + khalf;
    const float bias = bh[col];
#pragma unroll
    for (int j = 0; j < 8; ++j)
        oproj[(size_t)(rowbase + j) * NPROJ + col] = acc[j] + bias;
}

__global__ void k_lstm(const float* __restrict__ gates, float* __restrict__ c,
                       bf16_t* __restrict__ xh, float* __restrict__ out_t) {
    int idx = blockIdx.x * blockDim.x + threadIdx.x;
    if (idx >= BSZ * CDIM) return;
    int b = idx / CDIM, j = idx % CDIM;
    const float* g = gates + (size_t)b * G4;
    float gi = sigmoidf_(g[j]);
    float gf = sigmoidf_(g[CDIM + j]);
    float gg = tanhf(g[2 * CDIM + j]);
    float go = sigmoidf_(g[3 * CDIM + j]);
    float cn = gf * c[idx] + gi * gg;
    float h = go * tanhf(cn);
    c[idx] = cn;
    xh[(size_t)b * KTOT + (IDIM + MMEM) + j] = (bf16_t)h;
    out_t[(size_t)b * HOUT + j] = h;
}

// -------- addressing helpers (blockDim == 128 == NMEM) --------
DEV float blk_sum(float v, float* s_red) {
    int tid = threadIdx.x;
    s_red[tid] = v; __syncthreads();
#pragma unroll
    for (int off = 64; off > 0; off >>= 1) {
        if (tid < off) s_red[tid] += s_red[tid + off];
        __syncthreads();
    }
    float r = s_red[0]; __syncthreads();
    return r;
}
DEV float blk_max(float v, float* s_red) {
    int tid = threadIdx.x;
    s_red[tid] = v; __syncthreads();
#pragma unroll
    for (int off = 64; off > 0; off >>= 1) {
        if (tid < off) s_red[tid] = fmaxf(s_red[tid], s_red[tid + off]);
        __syncthreads();
    }
    float r = s_red[0]; __syncthreads();
    return r;
}

// One head's addressing; leaves normalized w in s_w and updates wprev.
DEV void address_head(const float* __restrict__ o, float* __restrict__ wprev,
                      const float* __restrict__ mem_b,
                      float* s_k, float* s_w, float* s_red) {
    const int tid = threadIdx.x;
    const float beta  = softplusf_(o[MMEM]);
    const float g     = sigmoidf_(o[MMEM + 1]);
    const float gamma = 1.0f + softplusf_(o[MMEM + 5]);
    // shift softmax (3 logits)
    float l0 = o[MMEM + 2], l1 = o[MMEM + 3], l2 = o[MMEM + 4];
    float lm = fmaxf(l0, fmaxf(l1, l2));
    float e0 = expf(l0 - lm), e1 = expf(l1 - lm), e2 = expf(l2 - lm);
    float es = e0 + e1 + e2;
    float sh0 = e0 / es, sh1 = e1 / es, sh2 = e2 / es;

    if (tid < MMEM) s_k[tid] = o[tid] + 1e-16f;
    __syncthreads();
    float pk = (tid < MMEM) ? s_k[tid] * s_k[tid] : 0.0f;
    float nk = fmaxf(sqrtf(blk_sum(pk, s_red)), 1e-8f);

    const float* row = mem_b + tid * MMEM;
    float dot = 0.0f, nm2 = 0.0f;
#pragma unroll 4
    for (int m = 0; m < MMEM; ++m) {
        float v = row[m] + 1e-16f;
        dot += v * s_k[m];
        nm2 += v * v;
    }
    float nm = fmaxf(sqrtf(nm2), 1e-8f);
    float sim = beta * (dot / (nm * nk));

    float mxs = blk_max(sim, s_red);
    float ex = expf(sim - mxs);
    float se = blk_sum(ex, s_red);
    float wc = ex / se;
    float wg = g * wc + (1.0f - g) * wprev[tid];
    s_w[tid] = wg; __syncthreads();
    float ww = sh0 * s_w[(tid + NMEM - 1) & (NMEM - 1)] + sh1 * wg +
               sh2 * s_w[(tid + 1) & (NMEM - 1)];
    __syncthreads();
    float w = powf(fmaxf(ww, 0.0f), gamma);
    float sw = blk_sum(w, s_red);
    w = w / (sw + 1e-16f);
    wprev[tid] = w;
    s_w[tid] = w;
    __syncthreads();
}

// per-batch block: read head -> read vector -> write head -> memory update
__global__ void k_ntm_heads(const float* __restrict__ oproj, float* __restrict__ mem,
                            float* __restrict__ wst, bf16_t* __restrict__ xh,
                            float* __restrict__ out_t) {
    __shared__ float s_k[MMEM];
    __shared__ float s_w[NMEM];
    __shared__ float s_red[NMEM];
    __shared__ float s_e[MMEM];
    __shared__ float s_a[MMEM];

    const int b = blockIdx.x;
    const int tid = threadIdx.x;
    const float* ob = oproj + (size_t)b * NPROJ;
    float* mem_b = mem + (size_t)b * NMEM * MMEM;

    // ---- read head ----
    address_head(ob, wst + (size_t)b * NMEM, mem_b, s_k, s_w, s_red);
    if (tid < MMEM) {
        float r = 0.0f;
#pragma unroll 4
        for (int n = 0; n < NMEM; ++n) r += s_w[n] * mem_b[n * MMEM + tid];
        out_t[(size_t)b * HOUT + CDIM + tid] = r;
        xh[(size_t)b * KTOT + IDIM + tid] = (bf16_t)r;
    }
    __syncthreads();

    // ---- write head ----
    const float* ow = ob + RPAD;
    address_head(ow, wst + (size_t)(BSZ + b) * NMEM, mem_b, s_k, s_w, s_red);
    if (tid < MMEM) {
        s_e[tid] = sigmoidf_(ow[MMEM + 6 + tid]);
        s_a[tid] = ow[2 * MMEM + 6 + tid];
    }
    __syncthreads();
    for (int idx = tid; idx < NMEM * MMEM; idx += blockDim.x) {
        int n = idx >> 6, m = idx & (MMEM - 1);
        float w = s_w[n];
        float old = mem_b[idx];
        mem_b[idx] = old * (1.0f - w * s_e[m]) + w * s_a[m];
    }
}

// ================= host side =================
extern "C" void kernel_launch(void* const* d_in, const int* in_sizes, int n_in,
                              void* d_out, int out_size, void* d_ws, size_t ws_size,
                              hipStream_t stream) {
    (void)in_sizes; (void)n_in; (void)out_size; (void)ws_size;
    const float* inp      = (const float*)d_in[0];
    const float* hid      = (const float*)d_in[1];
    const float* c0       = (const float*)d_in[2];
    const float* mem_bias = (const float*)d_in[3];
    const float* r0       = (const float*)d_in[4];
    const float* W_ih     = (const float*)d_in[5];
    const float* W_hh     = (const float*)d_in[6];
    const float* b_ih     = (const float*)d_in[7];
    const float* b_hh     = (const float*)d_in[8];
    const float* read_W   = (const float*)d_in[9];
    const float* read_b   = (const float*)d_in[10];
    const float* write_W  = (const float*)d_in[11];
    const float* write_b  = (const float*)d_in[12];
    float* out = (float*)d_out;

    // workspace layout (256B-aligned chunks)
    char* ws = (char*)d_ws;
    size_t off = 0;
    auto take = [&](size_t bytes) {
        size_t o = off;
        off = (off + bytes + 255) & ~(size_t)255;
        return o;
    };
    bf16_t* xh    = (bf16_t*)(ws + take((size_t)BSZ * KTOT * sizeof(bf16_t)));
    bf16_t* Wc    = (bf16_t*)(ws + take((size_t)G4 * KTOT * sizeof(bf16_t)));
    bf16_t* WH    = (bf16_t*)(ws + take((size_t)NPROJ * CDIM * sizeof(bf16_t)));
    float*  bc    = (float*)(ws + take((size_t)G4 * sizeof(float)));
    float*  bh    = (float*)(ws + take((size_t)NPROJ * sizeof(float)));
    float*  gates = (float*)(ws + take((size_t)BSZ * G4 * sizeof(float)));
    float*  oproj = (float*)(ws + take((size_t)BSZ * NPROJ * sizeof(float)));
    float*  cbuf  = (float*)(ws + take((size_t)BSZ * CDIM * sizeof(float)));
    float*  mem   = (float*)(ws + take((size_t)BSZ * NMEM * MMEM * sizeof(float)));
    float*  wst   = (float*)(ws + take((size_t)2 * BSZ * NMEM * sizeof(float)));

    // ---- one-time init (runs every call; deterministic) ----
    k_init_Wc<<<(G4 * KTOT + 255) / 256, 256, 0, stream>>>(W_ih, W_hh, Wc);
    k_init_WH<<<(NPROJ * CDIM + 255) / 256, 256, 0, stream>>>(read_W, write_W, WH);
    k_init_bias<<<(G4 + 255) / 256, 256, 0, stream>>>(b_ih, b_hh, read_b, write_b, bc, bh);
    k_init_mem<<<(BSZ * NMEM * MMEM + 255) / 256, 256, 0, stream>>>(mem_bias, mem);
    k_init_ch<<<(BSZ * CDIM + 255) / 256, 256, 0, stream>>>(hid, c0, cbuf, xh);
    k_init_reads<<<(BSZ * MMEM + 255) / 256, 256, 0, stream>>>(r0, xh);
    k_init_wst<<<(2 * BSZ * NMEM + 255) / 256, 256, 0, stream>>>(wst);

    // ---- recurrent steps ----
    for (int t = 0; t < T_STEPS; ++t) {
        const float* inp_t = inp + (size_t)t * BSZ * IDIM;
        float* out_t = out + (size_t)t * BSZ * HOUT;
        k_copy_inp<<<(BSZ * IDIM + 255) / 256, 256, 0, stream>>>(inp_t, xh);
        k_gemm_gates<<<dim3(32, 8), 256, 0, stream>>>(xh, Wc, bc, gates);
        k_lstm<<<(BSZ * CDIM + 255) / 256, 256, 0, stream>>>(gates, cbuf, xh, out_t);
        k_gemm_proj<<<dim3(3, 16), 192, 0, stream>>>(xh, WH, bh, oproj);
        k_ntm_heads<<<BSZ, NMEM, 0, stream>>>(oproj, mem, wst, xh, out_t);
    }
}
